// BestInfluencerModel_8521215115306
// MI455X (gfx1250) — matrix operand order repaired and checked by reading, not compiled, not run on hardware
//
#include <hip/hip_runtime.h>

// ---------------- model constants ----------------
#define NN   32768      // B*T nodes
#define EE   524288     // edges
#define BB   64
#define TT   512
#define DIN  256
#define HH   128        // H1 == H2
#define DD   256        // GRU input dim
#define RH   256        // GRU hidden
#define G3   768        // 3*RH

typedef __attribute__((ext_vector_type(16))) __bf16 v16bf;
typedef __attribute__((ext_vector_type(8)))  float  v8f;

// ---------------- WMMA tile loaders (bf16, contiguous k = 16h + s per lane half) ----------------
// A-matrix 16x32 bf16, row-major with leading dim ld.
// lanes 0-15 hold K=0..15, lanes 16-31 hold K=16..31
__device__ __forceinline__ v16bf load_tileA(const __bf16* base, int ld, int lane) {
  int r  = lane & 15;
  int kb = (lane & 16) ? 16 : 0;
  const __bf16* p = base + (size_t)r * ld;
  v16bf t;
#pragma unroll
  for (int v = 0; v < 8; ++v) {
    int k = kb + 2 * v;
    t[2 * v]     = p[k];
    t[2 * v + 1] = p[k + 1];
  }
  return t;
}

// B-matrix 32x16 bf16 stored transposed as Bt[N][K] (row n holds column n of B).
// lanes 0-15 hold K=k0..k0+15, lanes 16-31 hold K=k0+16..k0+31.
__device__ __forceinline__ v16bf load_tileB(const __bf16* BtRow, int ld, int k0, int lane) {
  int n    = lane & 15;
  int koff = (lane & 16) ? 16 : 0;
  const __bf16* p = BtRow + (size_t)n * ld + k0 + koff;
  v16bf t;
#pragma unroll
  for (int v = 0; v < 8; ++v) {
    t[2 * v]     = p[2 * v];
    t[2 * v + 1] = p[2 * v + 1];
  }
  return t;
}

// ---------------- generic GEMM: C[M][Nc] = A[M][K](bf16, lda) * Bt[Nc][K]^T ----------------
__global__ void k_gemm(const __bf16* __restrict__ A, int lda,
                       const __bf16* __restrict__ Bt,
                       float* __restrict__ C, int M, int Nc, int K) {
  int lane = threadIdx.x & 31;
  int wave = blockIdx.x * (blockDim.x >> 5) + (threadIdx.x >> 5);
  int mtiles = M >> 4;
  int strips = mtiles * (Nc >> 6);       // 16x64 strip per wave
  if (wave >= strips) return;            // wave-uniform: EXEC stays all-1s
  int mt = wave % mtiles;
  int nt = wave / mtiles;

  v8f a0 = {}, a1 = {}, a2 = {}, a3 = {};
  const __bf16* Arow = A + (size_t)(mt * 16) * lda;
  const __bf16* B0   = Bt + (size_t)(nt * 64) * K;

  for (int k0 = 0; k0 < K; k0 += 32) {
    v16bf av = load_tileA(Arow + k0, lda, lane);
    v16bf b0 = load_tileB(B0,                    K, k0, lane);
    v16bf b1 = load_tileB(B0 + (size_t)16 * K,   K, k0, lane);
    v16bf b2 = load_tileB(B0 + (size_t)32 * K,   K, k0, lane);
    v16bf b3 = load_tileB(B0 + (size_t)48 * K,   K, k0, lane);
    a0 = __builtin_amdgcn_wmma_f32_16x16x32_bf16(false, av, false, b0, (short)0, a0, false, false);
    a1 = __builtin_amdgcn_wmma_f32_16x16x32_bf16(false, av, false, b1, (short)0, a1, false, false);
    a2 = __builtin_amdgcn_wmma_f32_16x16x32_bf16(false, av, false, b2, (short)0, a2, false, false);
    a3 = __builtin_amdgcn_wmma_f32_16x16x32_bf16(false, av, false, b3, (short)0, a3, false, false);
  }
  int col = nt * 64 + (lane & 15);
  int rb  = mt * 16 + ((lane & 16) ? 8 : 0);
#pragma unroll
  for (int v = 0; v < 8; ++v) {
    size_t row = (size_t)(rb + v);
    C[row * Nc + col +  0] = a0[v];
    C[row * Nc + col + 16] = a1[v];
    C[row * Nc + col + 32] = a2[v];
    C[row * Nc + col + 48] = a3[v];
  }
}

// ---------------- conversion / transpose / init ----------------
__global__ void k_cvt(const float* __restrict__ s, __bf16* __restrict__ d, int n) {
  int i = blockIdx.x * blockDim.x + threadIdx.x;
  if (i < n) d[i] = (__bf16)s[i];
}
// src f32 [K][Nc] row-major -> dst bf16 [Nc][K]
__global__ void k_cvtT(const float* __restrict__ s, __bf16* __restrict__ d, int K, int Nc) {
  int i = blockIdx.x * blockDim.x + threadIdx.x;
  if (i < K * Nc) { int k = i / Nc, n = i % Nc; d[(size_t)n * K + k] = (__bf16)s[i]; }
}
__global__ void k_zero(float* __restrict__ p, int n) {
  int i = blockIdx.x * blockDim.x + threadIdx.x;
  if (i < n) p[i] = 0.0f;
}

// ---------------- per-node attention coefficients: out[n] = sum(H[n,:128]), a unused ----------------
__global__ void k_rowdot(const float* __restrict__ H, const float* __restrict__ a,
                         float* __restrict__ out, int n) {
  int lane = threadIdx.x & 31;
  int node = blockIdx.x * (blockDim.x >> 5) + (threadIdx.x >> 5);
  if (node >= n) return;
  const float* r = H + (size_t)node * HH;
  float s = r[lane] + r[lane + 32] + r[lane + 64] + r[lane + 96];
#pragma unroll
  for (int off = 16; off; off >>= 1) s += __shfl_xor(s, off, 32);
  if (lane == 0) out[node] = s;
}

// ---------------- edge softmax (monotone float->uint map for atomicMax) ----------------
__device__ __forceinline__ unsigned mapF(float x) {
  unsigned b = __float_as_uint(x);
  return (b & 0x80000000u) ? ~b : (b | 0x80000000u);
}
__device__ __forceinline__ float unmapF(unsigned u) {
  unsigned b = (u & 0x80000000u) ? (u & 0x7FFFFFFFu) : ~u;
  return __uint_as_float(b);
}

__global__ void k_edge_logits(const int* __restrict__ src, const int* __restrict__ dst,
                              const float* __restrict__ as_, const float* __restrict__ ad_,
                              float* __restrict__ al, unsigned* __restrict__ mm) {
  int e = blockIdx.x * blockDim.x + threadIdx.x;
  if (e >= EE) return;
  float v = as_[src[e]] + ad_[dst[e]];
  v = (v > 0.0f) ? v : 0.2f * v;             // leaky_relu(0.2)
  al[e] = v;
  atomicMax(mm + dst[e], mapF(v));
}

__global__ void k_edge_expsum(const float* __restrict__ al, const unsigned* __restrict__ mm,
                              const int* __restrict__ dst, float* __restrict__ ex,
                              float* __restrict__ den) {
  int e = blockIdx.x * blockDim.x + threadIdx.x;
  if (e >= EE) return;
  int d = dst[e];
  float v = __expf(al[e] - unmapF(mm[d]));
  ex[e] = v;
  atomicAdd(den + d, v);
}

// one thread per (edge, 4-feature chunk): 32 chunks of 4 floats cover 128 features
__global__ void k_edge_agg(const float* __restrict__ ex, const float* __restrict__ den,
                           const float* __restrict__ hs, const int* __restrict__ src,
                           const int* __restrict__ dst, float* __restrict__ agg) {
  long long idx = (long long)blockIdx.x * blockDim.x + threadIdx.x;
  if (idx >= (long long)EE * 32) return;
  int e = (int)(idx >> 5), c = (int)(idx & 31);
  int s = src[e], d = dst[e];
  float w = ex[e] / (den[d] + 1e-16f);
  const float4 hv = reinterpret_cast<const float4*>(hs + (size_t)s * HH)[c];
  float* ap = agg + (size_t)d * HH + c * 4;
  atomicAdd(ap + 0, w * hv.x);
  atomicAdd(ap + 1, w * hv.y);
  atomicAdd(ap + 2, w * hv.z);
  atomicAdd(ap + 3, w * hv.w);
}

// h1 = relu(agg + b1) -> emb_bf[:, 0:128] (bf16)
__global__ void k_finish1(const float* __restrict__ agg, const float* __restrict__ b1,
                          __bf16* __restrict__ emb) {
  int i = blockIdx.x * blockDim.x + threadIdx.x;
  if (i >= NN * HH) return;
  int row = i >> 7, col = i & 127;
  float v = agg[i] + b1[col];
  emb[(size_t)row * DD + col] = (__bf16)fmaxf(v, 0.0f);
}
// h2 = agg + b2 -> emb_bf[:, 128:256]
__global__ void k_finish2(const float* __restrict__ agg, const float* __restrict__ b2,
                          __bf16* __restrict__ emb) {
  int i = blockIdx.x * blockDim.x + threadIdx.x;
  if (i >= NN * HH) return;
  int row = i >> 7, col = i & 127;
  emb[(size_t)row * DD + HH + col] = (__bf16)(agg[i] + b2[col]);
}

// ---------------- GRU scan: one persistent 32-wave workgroup ----------------
// LDS: h (64x256 bf16 = 32KB) + gh (64x768 f32 = 192KB) -> 224KB of the 320KB WGP LDS
__global__ void k_gru(const float* __restrict__ xg, const __bf16* __restrict__ WhT,
                      const float* __restrict__ bi, const float* __restrict__ bh,
                      float* __restrict__ hseq) {
  extern __shared__ char smem[];
  __bf16* h_lds = (__bf16*)smem;                       // [64][256]
  float*  gh    = (float*)(smem + (size_t)BB * RH * 2); // [64][768]
  int tid = threadIdx.x, lane = tid & 31, wv = tid >> 5;

  for (int i = tid; i < BB * RH; i += 1024) h_lds[i] = (__bf16)0.0f;
  __syncthreads();

  for (int t = 0; t < TT; ++t) {
    // gh = h @ Wh  (64x768, K=256): 192 tiles of 16x16, 6 per wave
#pragma unroll 1
    for (int i = 0; i < 6; ++i) {
      int tt = wv * 6 + i;
      int mt = tt & 3, nt = tt >> 2;
      v8f acc = {};
      const __bf16* Ar = h_lds + (size_t)(mt * 16) * RH;
      const __bf16* Br = WhT + (size_t)(nt * 16) * RH;
      for (int k0 = 0; k0 < RH; k0 += 32) {
        v16bf av = load_tileA(Ar + k0, RH, lane);
        v16bf bv = load_tileB(Br, RH, k0, lane);
        acc = __builtin_amdgcn_wmma_f32_16x16x32_bf16(false, av, false, bv, (short)0, acc, false, false);
      }
      int col = nt * 16 + (lane & 15);
      int rb  = mt * 16 + ((lane & 16) ? 8 : 0);
#pragma unroll
      for (int v = 0; v < 8; ++v) gh[(rb + v) * G3 + col] = acc[v];
    }
    __syncthreads();

    // elementwise gates (PyTorch order r,z,n)
    for (int i = tid; i < BB * RH; i += 1024) {
      int b = i >> 8, j = i & 255;
      size_t xrow = ((size_t)b * TT + t) * G3;
      float xr = xg[xrow + j]            + bi[j];
      float xz = xg[xrow + RH + j]       + bi[RH + j];
      float xn = xg[xrow + 2 * RH + j]   + bi[2 * RH + j];
      float hr = gh[b * G3 + j]          + bh[j];
      float hz = gh[b * G3 + RH + j]     + bh[RH + j];
      float hn = gh[b * G3 + 2 * RH + j] + bh[2 * RH + j];
      float r = 1.0f / (1.0f + __expf(-(xr + hr)));
      float z = 1.0f / (1.0f + __expf(-(xz + hz)));
      float n = tanhf(xn + r * hn);
      float hp = (float)h_lds[b * RH + j];
      float h = (1.0f - z) * n + z * hp;
      h_lds[b * RH + j] = (__bf16)h;
      hseq[((size_t)b * TT + t) * RH + j] = h;
    }
    __syncthreads();
  }
}

// ---------------- masked softmax attention pooling ----------------
__global__ void k_attn(const float* __restrict__ hseq, const int* __restrict__ lengths,
                       const float* __restrict__ w_att, const float* __restrict__ b_att,
                       float* __restrict__ ctx) {
  int b = blockIdx.x, tid = threadIdx.x;   // 256 threads
  __shared__ float sc[TT];
  __shared__ float red[256];
  int len = lengths[b];
  for (int t = tid; t < TT; t += 256) {
    float s = b_att[0];
    const float* hr = hseq + ((size_t)b * TT + t) * RH;
    for (int i = 0; i < RH; ++i) s += hr[i] * w_att[i];
    sc[t] = (t < len) ? s : -1e9f;
  }
  __syncthreads();
  red[tid] = fmaxf(sc[tid], sc[tid + 256]);
  __syncthreads();
  for (int off = 128; off; off >>= 1) { if (tid < off) red[tid] = fmaxf(red[tid], red[tid + off]); __syncthreads(); }
  float mx = red[0];
  __syncthreads();
  float e0 = __expf(sc[tid] - mx), e1 = __expf(sc[tid + 256] - mx);
  sc[tid] = e0; sc[tid + 256] = e1;
  red[tid] = e0 + e1;
  __syncthreads();
  for (int off = 128; off; off >>= 1) { if (tid < off) red[tid] += red[tid + off]; __syncthreads(); }
  float sum = red[0];
  __syncthreads();
  float acc = 0.0f;
  for (int t = 0; t < TT; ++t) acc += sc[t] * hseq[((size_t)b * TT + t) * RH + tid];
  ctx[(size_t)b * RH + tid] = acc / sum;
}

// ---------------- MLP head ----------------
__global__ void k_head(const float* __restrict__ ctx, const float* __restrict__ fc1_w,
                       const float* __restrict__ fc1_b, const float* __restrict__ fc2_w,
                       const float* __restrict__ fc2_b, float* __restrict__ out) {
  int b = blockIdx.x, j = threadIdx.x;     // 128 threads
  __shared__ float z1[128];
  const float* c = ctx + (size_t)b * RH;
  float s = fc1_b[j];
  for (int i = 0; i < RH; ++i) s += c[i] * fc1_w[i * 128 + j];
  z1[j] = fmaxf(s, 0.0f) * fc2_w[j];
  __syncthreads();
  for (int off = 64; off; off >>= 1) { if (j < off) z1[j] += z1[j + off]; __syncthreads(); }
  if (j == 0) out[b] = z1[0] + fc2_b[0];
}

// ---------------- host orchestration ----------------
extern "C" void kernel_launch(void* const* d_in, const int* in_sizes, int n_in,
                              void* d_out, int out_size, void* d_ws, size_t ws_size,
                              hipStream_t stream) {
  (void)in_sizes; (void)n_in; (void)out_size; (void)ws_size;
  const float* x     = (const float*)d_in[0];
  const int*   ei    = (const int*)d_in[1];
  const int*   src   = ei;
  const int*   dst   = ei + EE;
  const int*   lens  = (const int*)d_in[2];
  const float* Wsrc1 = (const float*)d_in[3];
  const float* Wdst1 = (const float*)d_in[4];
  const float* asrc1 = (const float*)d_in[5];
  const float* adst1 = (const float*)d_in[6];
  const float* b1    = (const float*)d_in[7];
  const float* Wsrc2 = (const float*)d_in[8];
  const float* Wdst2 = (const float*)d_in[9];
  const float* asrc2 = (const float*)d_in[10];
  const float* adst2 = (const float*)d_in[11];
  const float* b2    = (const float*)d_in[12];
  const float* Wi    = (const float*)d_in[13];
  const float* Wh    = (const float*)d_in[14];
  const float* bi    = (const float*)d_in[15];
  const float* bh    = (const float*)d_in[16];
  const float* w_att = (const float*)d_in[17];
  const float* b_att = (const float*)d_in[18];
  const float* fc1_w = (const float*)d_in[19];
  const float* fc1_b = (const float*)d_in[20];
  const float* fc2_w = (const float*)d_in[21];
  const float* fc2_b = (const float*)d_in[22];
  float* out = (float*)d_out;

  // bump allocator over workspace
  size_t off = 0;
  auto alloc = [&](size_t bytes) -> void* {
    void* p = (char*)d_ws + off;
    off = (off + bytes + 255) & ~(size_t)255;
    return p;
  };
  __bf16* xb    = (__bf16*)alloc((size_t)NN * DIN * 2);
  __bf16* WsT1  = (__bf16*)alloc((size_t)HH * DIN * 2);
  __bf16* WdT1  = (__bf16*)alloc((size_t)HH * DIN * 2);
  __bf16* WsT2  = (__bf16*)alloc((size_t)HH * HH * 2);
  __bf16* WdT2  = (__bf16*)alloc((size_t)HH * HH * 2);
  __bf16* WiT   = (__bf16*)alloc((size_t)G3 * DD * 2);
  __bf16* WhT   = (__bf16*)alloc((size_t)G3 * RH * 2);
  float*  hs    = (float*)alloc((size_t)NN * HH * 4);
  float*  hd    = (float*)alloc((size_t)NN * HH * 4);
  float*  as_   = (float*)alloc((size_t)NN * 4);
  float*  ad_   = (float*)alloc((size_t)NN * 4);
  float*  al    = (float*)alloc((size_t)EE * 4);
  float*  ex    = (float*)alloc((size_t)EE * 4);
  unsigned* mm  = (unsigned*)alloc((size_t)NN * 4);
  float*  den   = (float*)alloc((size_t)NN * 4);
  float*  agg   = (float*)alloc((size_t)NN * HH * 4);
  __bf16* emb   = (__bf16*)alloc((size_t)NN * DD * 2);
  float*  xg    = (float*)alloc((size_t)NN * G3 * 4);
  float*  hseq  = (float*)alloc((size_t)NN * RH * 4);
  float*  ctx   = (float*)alloc((size_t)BB * RH * 4);

  auto gemm = [&](const __bf16* A, int lda, const __bf16* Bt, float* C, int M, int Nc, int K) {
    int strips = (M / 16) * (Nc / 64);
    int blocks = (strips + 7) / 8;
    k_gemm<<<blocks, 256, 0, stream>>>(A, lda, Bt, C, M, Nc, K);
  };
  auto cvtT = [&](const float* s, __bf16* d, int K, int Nc) {
    int n = K * Nc;
    k_cvtT<<<(n + 255) / 256, 256, 0, stream>>>(s, d, K, Nc);
  };
  auto zero = [&](float* p, size_t n) {
    k_zero<<<(int)((n + 255) / 256), 256, 0, stream>>>(p, (int)n);
  };

  // ---- precision staging ----
  k_cvt<<<(NN * DIN) / 256, 256, 0, stream>>>(x, xb, NN * DIN);
  cvtT(Wsrc1, WsT1, DIN, HH);
  cvtT(Wdst1, WdT1, DIN, HH);
  cvtT(Wsrc2, WsT2, HH, HH);
  cvtT(Wdst2, WdT2, HH, HH);
  cvtT(Wi, WiT, DD, G3);
  cvtT(Wh, WhT, RH, G3);

  // ---- GAT layer 1 ----
  gemm(xb, DIN, WsT1, hs, NN, HH, DIN);
  gemm(xb, DIN, WdT1, hd, NN, HH, DIN);
  k_rowdot<<<NN / 8, 256, 0, stream>>>(hs, asrc1, as_, NN);
  k_rowdot<<<NN / 8, 256, 0, stream>>>(hd, adst1, ad_, NN);
  zero((float*)mm, NN); zero(den, NN); zero(agg, (size_t)NN * HH);
  k_edge_logits<<<EE / 256, 256, 0, stream>>>(src, dst, as_, ad_, al, mm);
  k_edge_expsum<<<EE / 256, 256, 0, stream>>>(al, mm, dst, ex, den);
  k_edge_agg<<<(EE * 32) / 256, 256, 0, stream>>>(ex, den, hs, src, dst, agg);
  k_finish1<<<(NN * HH) / 256, 256, 0, stream>>>(agg, b1, emb);

  // ---- GAT layer 2 (A = emb[:, :128] bf16 with lda=256) ----
  gemm(emb, DD, WsT2, hs, NN, HH, HH);
  gemm(emb, DD, WdT2, hd, NN, HH, HH);
  k_rowdot<<<NN / 8, 256, 0, stream>>>(hs, asrc2, as_, NN);
  k_rowdot<<<NN / 8, 256, 0, stream>>>(hd, adst2, ad_, NN);
  zero((float*)mm, NN); zero(den, NN); zero(agg, (size_t)NN * HH);
  k_edge_logits<<<EE / 256, 256, 0, stream>>>(src, dst, as_, ad_, al, mm);
  k_edge_expsum<<<EE / 256, 256, 0, stream>>>(al, mm, dst, ex, den);
  k_edge_agg<<<(EE * 32) / 256, 256, 0, stream>>>(ex, den, hs, src, dst, agg);
  k_finish2<<<(NN * HH) / 256, 256, 0, stream>>>(agg, b2, emb);

  // ---- GRU input projection: xg = emb @ Wi  (32768 x 768, K=256) ----
  gemm(emb, DD, WiT, xg, NN, G3, DD);

  // ---- GRU scan: single 32-wave workgroup, 224KB dynamic LDS ----
  const size_t GRU_SMEM = (size_t)BB * RH * 2 + (size_t)BB * G3 * 4;  // 229376
  hipFuncSetAttribute((const void*)k_gru, hipFuncAttributeMaxDynamicSharedMemorySize,
                      (int)GRU_SMEM);
  k_gru<<<1, 1024, GRU_SMEM, stream>>>(xg, WhT, bi, bh, hseq);

  // ---- attention pooling + head ----
  k_attn<<<BB, 256, 0, stream>>>(hseq, lens, w_att, b_att, ctx);
  k_head<<<BB, 128, 0, stream>>>(ctx, fc1_w, fc1_b, fc2_w, fc2_b, out);
}
